// UnifiedJumpDiffusionSSM_29575144801153
// MI455X (gfx1250) — matrix operand order repaired and checked by reading, not compiled, not run on hardware
//
#include <hip/hip_runtime.h>

// ---------------------------------------------------------------------------
// UnifiedJumpDiffusionSSM for MI455X (gfx1250, wave32, WMMA)
//
// Sizes: B=16, L=1024, D_MODEL=256, D_STATE=16, N_LAYERS=2, D_CONT=32, D_EVENT=16
//
// Pipeline:
//   1) prep_weights : f32 -> f16 weight conversion (Wd/Wb1/Wb2/Wc), K-major rows
//   2) prep_input   : sign*log1p, x@ip_W.T + LN  -> xproj[t][b][d]   (parallel)
//   3) prep_events  : et@We.T, et@Wae.T, et@Wbe.T for both layers    (parallel)
//   4) scan         : single 1024-thread workgroup, sequential over L.
//        - per (t,layer): v_wmma_f32_16x16x32_f16 for xt@Wd.T (16 tiles x 8 k),
//          xt@Wb1.T (2 tiles), xt@Wc.T (1 tile), silu(..)@Wb2.T (1 tile)
//        - state h kept in VGPRs: 32 waves * 32 lanes * 128 f32 = full state
//        - per-step LN via LDS + shfl reductions
//   5) readout      : gelu(y_last@ro_W.T)+LN                          (parallel)
// ---------------------------------------------------------------------------

typedef _Float16 half16_t __attribute__((ext_vector_type(16)));
typedef _Float16 half8_t  __attribute__((ext_vector_type(8)));
typedef float    float8_t __attribute__((ext_vector_type(8)));

#define LSTEPS 1024
#define NB     16      // batch
#define DM     256     // d_model
#define DSTATE 16

// ---- LDS layout for scan kernel (bytes) ----
#define XT32_OFF   0        // 4096 f32
#define YZ_OFF     16384    // 4096 f32
#define DS_OFF     32768    // 4096 f32  (dpre then ds)
#define EA_OFF     49152    // 4096 f32
#define AB_OFF     65536    // 8192 f32  (A_base[l][d][s])
#define XT16_OFF   98304    // 4096 f16
#define SIL_OFF    106496   // 512 f16
#define BPRE_OFF   107520   // 512 f32
#define BT_OFF     109568   // 256 f32
#define CT_OFF     110592   // 256 f32
#define BD_OFF     111616   // 512 f32
#define BB1_OFF    113664   // 64 f32
#define BB2_OFF    113920   // 32 f32
#define DSK_OFF    114048   // 512 f32
#define LNG_OFF    116096   // 512 f32
#define LNB_OFF    118144   // 512 f32
#define STAT_OFF   120192   // 32 f32
#define SCAN_SMEM  120320

__device__ __forceinline__ half16_t make_frag(const _Float16* p0, const _Float16* p1) {
  half8_t lo = *(const half8_t*)p0;
  half8_t hi = *(const half8_t*)p1;
  half16_t r;
#pragma unroll
  for (int e = 0; e < 8; ++e) { r[e] = lo[e]; r[e + 8] = hi[e]; }
  return r;
}

__device__ __forceinline__ float8_t wmma16(half16_t a, half16_t b, float8_t c) {
  return __builtin_amdgcn_wmma_f32_16x16x32_f16(false, a, false, b, (short)0, c, false, false);
}

// ---------------------------------------------------------------------------
// 1) weight conversion to f16 (row-major [out][in], i.e. K-contiguous rows)
// ---------------------------------------------------------------------------
__global__ void prep_weights_kernel(const float* __restrict__ Wd,
                                    const float* __restrict__ Wb1,
                                    const float* __restrict__ Wb2,
                                    const float* __restrict__ Wc,
                                    _Float16* __restrict__ wd16,
                                    _Float16* __restrict__ wb116,
                                    _Float16* __restrict__ wb216,
                                    _Float16* __restrict__ wc16) {
  int i = blockIdx.x * blockDim.x + threadIdx.x;
  int stride = gridDim.x * blockDim.x;
  for (int k = i; k < 2 * DM * DM; k += stride) wd16[k]  = (_Float16)Wd[k];
  for (int k = i; k < 2 * 32 * DM; k += stride) wb116[k] = (_Float16)Wb1[k];
  for (int k = i; k < 2 * 16 * 32; k += stride) wb216[k] = (_Float16)Wb2[k];
  for (int k = i; k < 2 * 16 * DM; k += stride) wc16[k]  = (_Float16)Wc[k];
}

// ---------------------------------------------------------------------------
// 2) input projection + LN : xproj[(t*16+b)*256 + j]
// ---------------------------------------------------------------------------
__global__ __launch_bounds__(256) void prep_input_kernel(
    const float* __restrict__ x, const float* __restrict__ ipW,
    const float* __restrict__ ipb, const float* __restrict__ lg,
    const float* __restrict__ lb, float* __restrict__ xproj) {
  __shared__ float xcS[32];
  __shared__ float red1[8], red2[8], stat[2];
  int r = blockIdx.x;            // r = b*1024 + t
  int b = r >> 10, t = r & 1023;
  int j = threadIdx.x;
  if (j < 32) {
    float v = x[(long)r * 48 + j];
    xcS[j] = copysignf(log1pf(fabsf(v)), v);
  }
  __syncthreads();
  float acc = ipb[j];
#pragma unroll
  for (int k = 0; k < 32; ++k) acc = fmaf(xcS[k], ipW[j * 32 + k], acc);
  float s1 = acc, s2 = acc * acc;
#pragma unroll
  for (int o = 16; o > 0; o >>= 1) {
    s1 += __shfl_xor(s1, o, 32);
    s2 += __shfl_xor(s2, o, 32);
  }
  int wv = j >> 5;
  if ((j & 31) == 0) { red1[wv] = s1; red2[wv] = s2; }
  __syncthreads();
  if (j == 0) {
    float a1 = 0.f, a2 = 0.f;
#pragma unroll
    for (int w = 0; w < 8; ++w) { a1 += red1[w]; a2 += red2[w]; }
    float m = a1 * (1.f / 256.f);
    float var = a2 * (1.f / 256.f) - m * m;
    stat[0] = m;
    stat[1] = rsqrtf(var + 1e-5f);
  }
  __syncthreads();
  xproj[((long)t * NB + b) * DM + j] = (acc - stat[0]) * stat[1] * lg[j] + lb[j];
}

// ---------------------------------------------------------------------------
// 3) event projections: ed/ea[(l*1024+t)*4096 + b*256 + d], eb[(l*1024+t)*256+b*16+s]
// ---------------------------------------------------------------------------
__global__ __launch_bounds__(256) void prep_events_kernel(
    const float* __restrict__ x, const float* __restrict__ We,
    const float* __restrict__ Wae, const float* __restrict__ Wbe,
    float* __restrict__ ed, float* __restrict__ ea, float* __restrict__ eb) {
  __shared__ float etS[16];
  int r = blockIdx.x;            // r = b*1024 + t
  int b = r >> 10, t = r & 1023;
  int j = threadIdx.x;
  if (j < 16) etS[j] = x[(long)r * 48 + 32 + j];
  __syncthreads();
  for (int l = 0; l < 2; ++l) {
    float a1 = 0.f, a2 = 0.f;
#pragma unroll
    for (int k = 0; k < 16; ++k) {
      float e = etS[k];
      a1 = fmaf(e, We[(l * DM + j) * 16 + k], a1);
      a2 = fmaf(e, Wae[(l * DM + j) * 16 + k], a2);
    }
    long o = ((long)(l * LSTEPS + t) * NB + b) * DM + j;
    ed[o] = a1;
    ea[o] = a2;
    if (j < 16) {
      float a3 = 0.f;
#pragma unroll
      for (int k = 0; k < 16; ++k) a3 = fmaf(etS[k], Wbe[(l * 16 + j) * 16 + k], a3);
      eb[(long)(l * LSTEPS + t) * 256 + b * 16 + j] = a3;
    }
  }
}

// ---------------------------------------------------------------------------
// 4) sequential scan: 1 workgroup, 1024 threads (32 waves)
// ---------------------------------------------------------------------------
__global__ __launch_bounds__(1024) void scan_kernel(
    const float* __restrict__ xproj, const float* __restrict__ ed,
    const float* __restrict__ ea, const float* __restrict__ eb,
    const _Float16* __restrict__ wd16, const _Float16* __restrict__ wb116,
    const _Float16* __restrict__ wb216, const _Float16* __restrict__ wc16,
    const float* __restrict__ A_log, const float* __restrict__ bd,
    const float* __restrict__ bb1, const float* __restrict__ bb2,
    const float* __restrict__ Dsk, const float* __restrict__ lng,
    const float* __restrict__ lnb, float* __restrict__ ylast) {
  extern __shared__ unsigned char smem[];
  float* xt32     = (float*)(smem + XT32_OFF);
  float* yz       = (float*)(smem + YZ_OFF);
  float* dsS      = (float*)(smem + DS_OFF);
  float* eaS      = (float*)(smem + EA_OFF);
  float* abS      = (float*)(smem + AB_OFF);
  _Float16* xt16  = (_Float16*)(smem + XT16_OFF);
  _Float16* sil16 = (_Float16*)(smem + SIL_OFF);
  float* bpreS    = (float*)(smem + BPRE_OFF);
  float* btS      = (float*)(smem + BT_OFF);
  float* ctS      = (float*)(smem + CT_OFF);
  float* bdS      = (float*)(smem + BD_OFF);
  float* bb1S     = (float*)(smem + BB1_OFF);
  float* bb2S     = (float*)(smem + BB2_OFF);
  float* dskS     = (float*)(smem + DSK_OFF);
  float* lngS     = (float*)(smem + LNG_OFF);
  float* lnbS     = (float*)(smem + LNB_OFF);
  float* statS    = (float*)(smem + STAT_OFF);

  const int tid = threadIdx.x;
  const int lane = tid & 31;
  const int wave = tid >> 5;
  const int waveHalf = wave >> 4;  // layer owned by this wave
  const int g = wave & 15;         // channel-group / tile id
  const int b_l = lane & 15;       // batch index for this lane
  const int sh = lane >> 4;        // state-half (s in [8*sh, 8*sh+8))
  const int koff = sh ? 16 : 0;    // WMMA A-fragment K offset per lane-half

  // one-time LDS init
  for (int i = tid; i < 2 * DM * DSTATE; i += 1024) {
    float a = A_log[i];
    a = fminf(fmaxf(a, -3.f), 1.f);
    abS[i] = -__expf(a);
  }
  for (int i = tid; i < 2 * DM; i += 1024) {
    bdS[i] = bd[i]; dskS[i] = Dsk[i]; lngS[i] = lng[i]; lnbS[i] = lnb[i];
  }
  if (tid < 64) bb1S[tid] = bb1[tid];
  if (tid < 32) bb2S[tid] = bb2[tid];

  // register-resident hidden state: h[layer=waveHalf][b=b_l][d=g*16+dd][s=sh*8+sl]
  float h[128];
#pragma unroll
  for (int i = 0; i < 128; ++i) h[i] = 0.f;

  __syncthreads();

  for (int t = 0; t < LSTEPS; ++t) {
    // load xt for this step (xt resets to xproj[t]; only h carries over)
    for (int i = tid; i < 4096; i += 1024) {
      float v = xproj[(long)t * 4096 + i];
      xt32[i] = v;
      xt16[i] = (_Float16)v;
    }
    __syncthreads();

    for (int l = 0; l < 2; ++l) {
      // ---- P1: WMMA matmuls ----
      if (waveHalf == l) {
        // delta_pre tile g : (16x256)@(256x256).T -> columns [16g,16g+16)
        float8_t c = {0.f, 0.f, 0.f, 0.f, 0.f, 0.f, 0.f, 0.f};
        const _Float16* wrow = wd16 + (long)l * DM * DM + (g * 16 + b_l) * DM + (sh ? 16 : 0);
#pragma unroll
        for (int k = 0; k < 8; ++k) {
          const _Float16* ab = xt16 + b_l * DM + k * 32 + koff;
          half16_t af = make_frag(ab, ab + 8);
          half16_t bf = make_frag(wrow + k * 32, wrow + k * 32 + 8);
          c = wmma16(af, bf, c);
        }
#pragma unroll
        for (int r = 0; r < 8; ++r) dsS[(r + sh * 8) * DM + g * 16 + b_l] = c[r];
      } else {
        if (g < 2) {
          // bpre tiles: (16x256)@(256x32).T
          float8_t c = {0.f, 0.f, 0.f, 0.f, 0.f, 0.f, 0.f, 0.f};
          const _Float16* wrow = wb116 + (long)l * 32 * DM + (g * 16 + b_l) * DM + (sh ? 16 : 0);
#pragma unroll
          for (int k = 0; k < 8; ++k) {
            const _Float16* ab = xt16 + b_l * DM + k * 32 + koff;
            half16_t af = make_frag(ab, ab + 8);
            half16_t bf = make_frag(wrow + k * 32, wrow + k * 32 + 8);
            c = wmma16(af, bf, c);
          }
#pragma unroll
          for (int r = 0; r < 8; ++r) bpreS[(r + sh * 8) * 32 + g * 16 + b_l] = c[r];
        } else if (g == 2) {
          // Ct: (16x256)@(256x16).T
          float8_t c = {0.f, 0.f, 0.f, 0.f, 0.f, 0.f, 0.f, 0.f};
          const _Float16* wrow = wc16 + (long)l * 16 * DM + b_l * DM + (sh ? 16 : 0);
#pragma unroll
          for (int k = 0; k < 8; ++k) {
            const _Float16* ab = xt16 + b_l * DM + k * 32 + koff;
            half16_t af = make_frag(ab, ab + 8);
            half16_t bf = make_frag(wrow + k * 32, wrow + k * 32 + 8);
            c = wmma16(af, bf, c);
          }
#pragma unroll
          for (int r = 0; r < 8; ++r) ctS[(r + sh * 8) * 16 + b_l] = c[r];
        } else if (g == 3) {
          // idle wave: prefetch next step's xt stream into caches
          const float* nxt = xproj + (long)((t + 1) & (LSTEPS - 1)) * 4096;
          __builtin_prefetch(nxt + lane * 32, 0, 1);
        }
      }
      __syncthreads();

      // ---- P2: ds = min(softplus(dpre + e_d + bd), 2); stage e_a; silu(bpre) ----
      {
        const float* edt = ed + (long)(l * LSTEPS + t) * 4096;
        const float* eat = ea + (long)(l * LSTEPS + t) * 4096;
        for (int i = tid; i < 4096; i += 1024) {
          int d = i & 255;
          float z = dsS[i] + edt[i] + bdS[l * DM + d];
          float sp = (z > 20.f) ? z : log1pf(__expf(z));
          dsS[i] = fminf(sp, 2.f);
          eaS[i] = eat[i];
        }
        if (tid < 512) {
          int j = tid & 31;
          float s = bpreS[tid] + bb1S[l * 32 + j];
          float sg = 1.f / (1.f + __expf(-s));
          sil16[tid] = (_Float16)(s * sg);
        }
      }
      __syncthreads();

      // ---- P3: Bt = silu(..) @ Wb2.T (one aux wave) ----
      if (wave == ((1 - l) << 4)) {
        const _Float16* ab = sil16 + b_l * 32 + koff;
        half16_t af = make_frag(ab, ab + 8);
        const _Float16* wrow = wb216 + l * 16 * 32 + b_l * 32 + (sh ? 16 : 0);
        half16_t bf = make_frag(wrow, wrow + 8);
        float8_t c = {0.f, 0.f, 0.f, 0.f, 0.f, 0.f, 0.f, 0.f};
        c = wmma16(af, bf, c);
#pragma unroll
        for (int r = 0; r < 8; ++r) btS[(r + sh * 8) * 16 + b_l] = c[r];
      }
      __syncthreads();

      // ---- P3b: Bt += bb2 + e_b ----
      if (tid < 256) {
        btS[tid] += bb2S[l * 16 + (tid & 15)] + eb[(long)(l * LSTEPS + t) * 256 + tid];
      }
      __syncthreads();

      // ---- P4: recurrence + y (owner waves only) ----
      if (waveHalf == l) {
#pragma unroll
        for (int dd = 0; dd < 16; ++dd) {
          int d = g * 16 + dd;
          float dsv = dsS[b_l * DM + d];
          float eav = eaS[b_l * DM + d];
          float xv = xt32[b_l * DM + d];
          float ys = 0.f;
#pragma unroll
          for (int sl = 0; sl < 8; ++sl) {
            int s = sh * 8 + sl;
            float abar = __expf(dsv * (abS[(l * DM + d) * DSTATE + s] + eav));
            float hv = h[dd * 8 + sl];
            hv = abar * hv + dsv * btS[b_l * 16 + s] * xv;
            h[dd * 8 + sl] = hv;
            ys = fmaf(hv, ctS[b_l * 16 + s], ys);
          }
          ys += __shfl_xor(ys, 16, 32);   // combine the two state halves
          if (sh == 0) yz[b_l * DM + d] = ys + dskS[l * DM + d] * xv + xv;  // y + residual
        }
      }
      __syncthreads();

      // ---- P5: LN stats (wave r handles batch row r) ----
      if (wave < 16) {
        float s1 = 0.f, s2 = 0.f;
#pragma unroll
        for (int jj = 0; jj < 8; ++jj) {
          float v = yz[wave * DM + jj * 32 + lane];
          s1 += v;
          s2 += v * v;
        }
#pragma unroll
        for (int o = 16; o > 0; o >>= 1) {
          s1 += __shfl_xor(s1, o, 32);
          s2 += __shfl_xor(s2, o, 32);
        }
        if (lane == 0) {
          float m = s1 * (1.f / 256.f);
          float var = s2 * (1.f / 256.f) - m * m;
          statS[wave * 2] = m;
          statS[wave * 2 + 1] = rsqrtf(var + 1e-5f);
        }
      }
      __syncthreads();

      // ---- P5c: normalize -> new xt ----
      for (int i = tid; i < 4096; i += 1024) {
        int b = i >> 8, d = i & 255;
        float xn = (yz[i] - statS[b * 2]) * statS[b * 2 + 1] * lngS[l * DM + d] + lnbS[l * DM + d];
        xt32[i] = xn;
        xt16[i] = (_Float16)xn;
      }
      __syncthreads();
    }  // layers

    if (t == LSTEPS - 1) {
      for (int i = tid; i < 4096; i += 1024) ylast[i] = xt32[i];
    }
  }
}

// ---------------------------------------------------------------------------
// 5) readout: LN(gelu(y_last @ ro_W.T + ro_b))
// ---------------------------------------------------------------------------
__global__ __launch_bounds__(256) void readout_kernel(
    const float* __restrict__ ylast, const float* __restrict__ roW,
    const float* __restrict__ rob, const float* __restrict__ rg,
    const float* __restrict__ rb2, float* __restrict__ out) {
  __shared__ float yS[256];
  __shared__ float red1[8], red2[8], stat[2];
  int b = blockIdx.x;
  int o = threadIdx.x;
  yS[o] = ylast[b * 256 + o];
  __syncthreads();
  float acc = rob[o];
  for (int k = 0; k < 256; ++k) acc = fmaf(yS[k], roW[o * 256 + k], acc);
  float gl = 0.5f * acc * (1.f + erff(acc * 0.7071067811865476f));
  float s1 = gl, s2 = gl * gl;
#pragma unroll
  for (int off = 16; off > 0; off >>= 1) {
    s1 += __shfl_xor(s1, off, 32);
    s2 += __shfl_xor(s2, off, 32);
  }
  int wv = o >> 5;
  if ((o & 31) == 0) { red1[wv] = s1; red2[wv] = s2; }
  __syncthreads();
  if (o == 0) {
    float a1 = 0.f, a2 = 0.f;
#pragma unroll
    for (int w = 0; w < 8; ++w) { a1 += red1[w]; a2 += red2[w]; }
    float m = a1 * (1.f / 256.f);
    float var = a2 * (1.f / 256.f) - m * m;
    stat[0] = m;
    stat[1] = rsqrtf(var + 1e-5f);
  }
  __syncthreads();
  out[b * 256 + o] = (gl - stat[0]) * stat[1] * rg[o] + rb2[o];
}

// ---------------------------------------------------------------------------
extern "C" void kernel_launch(void* const* d_in, const int* in_sizes, int n_in,
                              void* d_out, int out_size, void* d_ws, size_t ws_size,
                              hipStream_t stream) {
  (void)in_sizes; (void)n_in; (void)out_size; (void)ws_size;
  const float* x    = (const float*)d_in[0];
  const float* ipW  = (const float*)d_in[1];
  const float* ipb  = (const float*)d_in[2];
  const float* iplg = (const float*)d_in[3];
  const float* iplb = (const float*)d_in[4];
  const float* Alog = (const float*)d_in[5];
  const float* Dsk  = (const float*)d_in[6];
  const float* Wd   = (const float*)d_in[7];
  const float* bd   = (const float*)d_in[8];
  const float* We   = (const float*)d_in[9];
  const float* Wb1  = (const float*)d_in[10];
  const float* bb1  = (const float*)d_in[11];
  const float* Wb2  = (const float*)d_in[12];
  const float* bb2  = (const float*)d_in[13];
  const float* Wbe  = (const float*)d_in[14];
  const float* Wae  = (const float*)d_in[15];
  const float* Wc   = (const float*)d_in[16];
  const float* lng  = (const float*)d_in[17];
  const float* lnb  = (const float*)d_in[18];
  const float* roW  = (const float*)d_in[19];
  const float* rob  = (const float*)d_in[20];
  const float* rog  = (const float*)d_in[21];
  const float* rolb = (const float*)d_in[22];
  float* out = (float*)d_out;

  unsigned char* ws = (unsigned char*)d_ws;
  size_t off = 0;
  auto take = [&](size_t bytes) -> void* {
    void* p = ws + off;
    off += (bytes + 255) & ~(size_t)255;
    return p;
  };
  float*    xproj = (float*)take((size_t)16384 * 256 * 4);       // 16 MB
  float*    ed    = (float*)take((size_t)2 * 1024 * 4096 * 4);   // 32 MB
  float*    ea    = (float*)take((size_t)2 * 1024 * 4096 * 4);   // 32 MB
  float*    eb    = (float*)take((size_t)2 * 1024 * 256 * 4);    // 2 MB
  _Float16* wd16  = (_Float16*)take((size_t)131072 * 2);
  _Float16* wb116 = (_Float16*)take((size_t)16384 * 2);
  _Float16* wb216 = (_Float16*)take((size_t)1024 * 2);
  _Float16* wc16  = (_Float16*)take((size_t)8192 * 2);
  float*    ylast = (float*)take((size_t)4096 * 4);

  prep_weights_kernel<<<64, 256, 0, stream>>>(Wd, Wb1, Wb2, Wc, wd16, wb116, wb216, wc16);
  prep_input_kernel<<<16384, 256, 0, stream>>>(x, ipW, ipb, iplg, iplb, xproj);
  prep_events_kernel<<<16384, 256, 0, stream>>>(x, We, Wae, Wbe, ed, ea, eb);

  (void)hipFuncSetAttribute((const void*)scan_kernel,
                            hipFuncAttributeMaxDynamicSharedMemorySize, SCAN_SMEM);
  scan_kernel<<<1, 1024, SCAN_SMEM, stream>>>(xproj, ed, ea, eb, wd16, wb116, wb216,
                                              wc16, Alog, bd, bb1, bb2, Dsk, lng, lnb,
                                              ylast);
  readout_kernel<<<16, 256, 0, stream>>>(ylast, roW, rob, rog, rolb, out);
}
